// CIGA_12025908429177
// MI455X (gfx1250) — matrix operand order repaired and checked by reading, not compiled, not run on hardware
//
#include <hip/hip_runtime.h>
#include <hip/hip_bf16.h>

#define BATCH 256
#define NPER  512
#define EPER  4096
#define DIM   64
#define KKEEP 1024

typedef __attribute__((ext_vector_type(16))) _Float16 v16h;
typedef __attribute__((ext_vector_type(8)))  float    v8f;

// ---------------------------------------------------------------------------
// Kernel 1: fused edge-MLP scores via WMMA f16 (f32 accumulate).
// One workgroup per batch (8 wave32). LDS layout (dynamic, 133120 B):
//   [0      ) h slab  f16 : 512*64
//   [65536  ) W1      f16 : 128*256
//   [131072 ) b1      f32 : 256
//   [132096 ) W2      f32 : 256
// Each wave processes PAIRS of 16-edge M-tiles: A fragments (8 total, 64
// VGPRs) are register-resident; B fragments stream from LDS and each feeds
// two WMMAs. nt loop is kept rolled to bound register pressure (no spills).
// ---------------------------------------------------------------------------
__global__ __launch_bounds__(256) void mlp_scores(
    const float* __restrict__ h, const float* __restrict__ W1,
    const float* __restrict__ b1, const float* __restrict__ W2,
    const float* __restrict__ b2, const int* __restrict__ ei0,
    const int* __restrict__ ei1, float* __restrict__ scores)
{
    extern __shared__ char smem[];
    _Float16* hS  = (_Float16*)(smem);
    _Float16* w1S = (_Float16*)(smem + 65536);
    float*    b1S = (float*)(smem + 131072);
    float*    w2S = (float*)(smem + 132096);

    const int b   = blockIdx.x;
    const int tid = threadIdx.x;

    // Stage: h slab and W1 converted fp32 -> fp16, biases/W2 as fp32.
    const float* hsrc = h + (size_t)b * NPER * DIM;
    for (int i = tid; i < NPER * DIM; i += 256) hS[i]  = (_Float16)hsrc[i];
    for (int i = tid; i < 128 * 256;  i += 256) w1S[(i & 255) * 128 + (i >> 8)] = (_Float16)W1[i];
    if (tid < 256) { b1S[tid] = b1[tid]; w2S[tid] = W2[tid]; }
    __syncthreads();

    const int wave = tid >> 5;
    const int lane = tid & 31;
    const int m    = lane & 15;   // A row (edge in tile) / C column (N)
    const int kh   = lane >> 4;   // half-wave K sub-group
    const float b2v = b2[0];

    // B fragment base: lane holds W1^T row (N = m), elements = K values (8*kh + s, 16 + 8*kh + s).
    const _Float16* w1row = w1S + m * 128 + kh * 8;

    union U { v16h v; uint4 q[2]; };

    // 256 M-tiles / 8 waves = 32 per wave, processed as 16 pairs (uniform
    // trip counts -> EXEC stays all-ones around WMMA).
    for (int p = 0; p < 16; ++p) {
        const int mt0 = wave + 8 * (2 * p);
        const int mt1 = wave + 8 * (2 * p + 1);

        const int g0 = b * EPER + mt0 * 16 + m;
        const int g1 = b * EPER + mt1 * 16 + m;
        int rowl[2], coll[2];
        rowl[0] = ei0[g0] - b * NPER;  coll[0] = ei1[g0] - b * NPER;
        rowl[1] = ei0[g1] - b * NPER;  coll[1] = ei1[g1] - b * NPER;

        // Build the 8 A fragments (register-resident for whole nt loop).
        U A[2][4];
#pragma unroll
        for (int t = 0; t < 2; ++t) {
#pragma unroll
            for (int ks = 0; ks < 4; ++ks) {
                const int kbase = ks * 32;
                // concat(h[row], h[col]) split lands on K-step boundaries
                const int node = (kbase < 64) ? rowl[t] : coll[t];
                const int koff = (kbase & 63) + kh * 8;
                const _Float16* ap = hS + node * DIM + koff;
                A[t][ks].q[0] = *(const uint4*)(ap);       // ds_load_b128
                A[t][ks].q[1] = *(const uint4*)(ap + 16);  // ds_load_b128
            }
        }

        float acc0[8] = {0.f, 0.f, 0.f, 0.f, 0.f, 0.f, 0.f, 0.f};
        float acc1[8] = {0.f, 0.f, 0.f, 0.f, 0.f, 0.f, 0.f, 0.f};

#pragma unroll 1   // keep rolled: bounds register pressure (no scratch spills)
        for (int nt = 0; nt < 16; ++nt) {
            v8f c0 = {}, c1 = {};
#pragma unroll
            for (int ks = 0; ks < 4; ++ks) {
                U Bv;
                const _Float16* bp = w1row + nt * 16 * 128 + ks * 32;
                Bv.q[0] = *(const uint4*)(bp);       // ds_load_b128
                Bv.q[1] = *(const uint4*)(bp + 16);  // ds_load_b128
                c0 = __builtin_amdgcn_wmma_f32_16x16x32_f16(
                        false, A[0][ks].v, false, Bv.v, (short)0, c0, false, false);
                c1 = __builtin_amdgcn_wmma_f32_16x16x32_f16(
                        false, A[1][ks].v, false, Bv.v, (short)0, c1, false, false);
            }
            // Fused layer 2: relu(z1 + b1) * W2, accumulated over N tiles.
            const int   n    = nt * 16 + m;   // C layout: N = lane&15
            const float bias = b1S[n];
            const float w2v  = w2S[n];
#pragma unroll
            for (int r = 0; r < 8; ++r) {     // C layout: M = r + 8*kh
                float z0 = c0[r] + bias;  z0 = z0 > 0.f ? z0 : 0.f;
                float z1 = c1[r] + bias;  z1 = z1 > 0.f ? z1 : 0.f;
                acc0[r] += z0 * w2v;
                acc1[r] += z1 * w2v;
            }
        }

        // Reduce across the 16 N-lanes of each half-wave (xor 1,2,4,8 stay
        // inside [0,15] / [16,31]).
#pragma unroll
        for (int r = 0; r < 8; ++r) {
            float v0 = acc0[r];
            v0 += __shfl_xor(v0, 1);
            v0 += __shfl_xor(v0, 2);
            v0 += __shfl_xor(v0, 4);
            v0 += __shfl_xor(v0, 8);
            acc0[r] = v0;
            float v1 = acc1[r];
            v1 += __shfl_xor(v1, 1);
            v1 += __shfl_xor(v1, 2);
            v1 += __shfl_xor(v1, 4);
            v1 += __shfl_xor(v1, 8);
            acc1[r] = v1;
        }
        if (m == 0) {
#pragma unroll
            for (int r = 0; r < 8; ++r) {
                scores[b * EPER + mt0 * 16 + (r + 8 * kh)] = acc0[r] + b2v;
                scores[b * EPER + mt1 * 16 + (r + 8 * kh)] = acc1[r] + b2v;
            }
        }
    }
}

// ---------------------------------------------------------------------------
// Kernel 2: per-batch bitonic full sort (descending score, ascending index on
// ties -> matches stable argsort of -s), emit top-K weights, rest negated,
// and set node mask from kept edges.
// ---------------------------------------------------------------------------
__global__ __launch_bounds__(256) void topk_sort(
    const float* __restrict__ scores, const int* __restrict__ ei0,
    const int* __restrict__ ei1, float* __restrict__ out_cw,
    float* __restrict__ out_spu, unsigned* __restrict__ mask)
{
    __shared__ float sS[EPER];
    __shared__ int   sI[EPER];
    const int b = blockIdx.x, tid = threadIdx.x;

    for (int i = tid; i < EPER; i += 256) { sS[i] = scores[b * EPER + i]; sI[i] = i; }

    for (int k = 2; k <= EPER; k <<= 1) {
        for (int j = k >> 1; j > 0; j >>= 1) {
            __syncthreads();
            for (int i = tid; i < EPER; i += 256) {
                if ((i & j) == 0) {
                    const int p = i | j;
                    float s0 = sS[i], s1 = sS[p];
                    int   i0 = sI[i], i1 = sI[p];
                    // "i ranks before p" in desired global (descending) order
                    bool before = (s0 > s1) || (s0 == s1 && i0 < i1);
                    bool up     = ((i & k) == 0);
                    if (up != before) { sS[i] = s1; sS[p] = s0; sI[i] = i1; sI[p] = i0; }
                }
            }
        }
    }
    __syncthreads();

    for (int i = tid; i < EPER; i += 256) {
        const float sc  = sS[i];
        const int   idx = sI[i];
        const int   g   = b * EPER + idx;
        if (i < KKEEP) {
            out_cw[b * KKEEP + i] = sc;
            mask[ei0[g]] = 1u;              // benign write races
            mask[ei1[g]] = 1u;
        } else {
            out_spu[b * (EPER - KKEEP) + (i - KKEEP)] = -sc;
        }
    }
}

// ---------------------------------------------------------------------------
// Kernel 3: causal_rep[b, d] = sum over masked nodes in batch of h[n, d].
// ---------------------------------------------------------------------------
__global__ __launch_bounds__(256) void causal_rep_k(
    const float* __restrict__ h, const unsigned* __restrict__ mask,
    float* __restrict__ out_rep)
{
    __shared__ float red[4][DIM];
    const int b = blockIdx.x, tid = threadIdx.x;
    const int d = tid & 63, c = tid >> 6;
    float acc = 0.f;
    for (int n = c * 128; n < (c + 1) * 128; ++n) {
        const int node = b * NPER + n;
        if (mask[node]) acc += h[(size_t)node * DIM + d];
    }
    red[c][d] = acc;
    __syncthreads();
    if (c == 0)
        out_rep[b * DIM + d] = red[0][d] + red[1][d] + red[2][d] + red[3][d];
}

__global__ void zero_mask(unsigned* __restrict__ mask, int n)
{
    const int i = blockIdx.x * blockDim.x + threadIdx.x;
    if (i < n) mask[i] = 0u;
}

// ---------------------------------------------------------------------------
extern "C" void kernel_launch(void* const* d_in, const int* in_sizes, int n_in,
                              void* d_out, int out_size, void* d_ws, size_t ws_size,
                              hipStream_t stream)
{
    const float* h  = (const float*)d_in[0];
    const float* W1 = (const float*)d_in[1];
    const float* b1 = (const float*)d_in[2];
    const float* W2 = (const float*)d_in[3];
    const float* b2 = (const float*)d_in[4];
    const int*   ei = (const int*)d_in[5];
    const int* ei0 = ei;
    const int* ei1 = ei + (size_t)BATCH * EPER;

    float* out      = (float*)d_out;
    float* out_rep  = out;                                   // 256*64
    float* out_cw   = out + BATCH * DIM;                     // 256*1024
    float* out_spu  = out + BATCH * DIM + BATCH * KKEEP;     // 256*3072

    float*    scores = (float*)d_ws;                                     // 4 MB
    unsigned* mask   = (unsigned*)((char*)d_ws +
                                   (size_t)BATCH * EPER * sizeof(float)); // 512 KB

    zero_mask<<<(BATCH * NPER + 255) / 256, 256, 0, stream>>>(mask, BATCH * NPER);
    mlp_scores<<<BATCH, 256, 133120, stream>>>(h, W1, b1, W2, b2, ei0, ei1, scores);
    topk_sort<<<BATCH, 256, 0, stream>>>(scores, ei0, ei1, out_cw, out_spu, mask);
    causal_rep_k<<<BATCH, 256, 0, stream>>>(h, mask, out_rep);
}
